// ImprovedMultitaskHeads_23493471109271
// MI455X (gfx1250) — matrix operand order repaired and checked by reading, not compiled, not run on hardware
//
#include <hip/hip_runtime.h>
#include <hip/hip_bf16.h>

// ---------------------------------------------------------------------------
// ImprovedMultitaskHeads: routed (grouped-GEMM) multitask adapter on gfx1250.
// B=32, S=256, D=512, N=16 heads, H=256, O=2, VOCAB=1024.
// Route tokens to heads, bucket into 16-row tiles, run the two big layers as
// f16 WMMA GEMMs (v_wmma_f32_16x16x32_f16, f32 accumulate). A-fragments are
// loaded once per k-step and shared across two N-tile accumulators.
// ---------------------------------------------------------------------------

typedef _Float16 v16h __attribute__((ext_vector_type(16)));
typedef _Float16 v8h  __attribute__((ext_vector_type(8)));
typedef float    v8f  __attribute__((ext_vector_type(8)));

#define NTOK    8192          // B*S
#define DDIM    512
#define HDIM    256
#define NHEAD   16
#define VOCAB_N 1024
#define LISTLEN 8448          // 8192 + 16*16 padding headroom (>= 8432 worst case)
#define MTILES  (LISTLEN/16)  // 528

// ---------------- routing ----------------

__global__ void k_init_counts(int* __restrict__ counts, int* __restrict__ cursors) {
    int t = threadIdx.x;
    if (t < NHEAD) { counts[t] = 0; cursors[t] = 0; }
}

__global__ void k_route(const int* __restrict__ tasks, const int* __restrict__ lut,
                        int* __restrict__ head, int* __restrict__ counts,
                        int* __restrict__ list, float* __restrict__ out) {
    int t = blockIdx.x * blockDim.x + threadIdx.x;
    if (t < LISTLEN) list[t] = -1;           // pad marker for unfilled slots
    if (t >= NTOK) return;
    out[2 * t + 0] = 0.0f;                   // invalid tokens must output zeros
    out[2 * t + 1] = 0.0f;
    int task = tasks[t];
    int h = -1;
    if (task >= 0 && task < VOCAB_N) {
        h = lut[task];
        if (h < 0 || h >= NHEAD) h = -1;
    }
    head[t] = h;
    if (h >= 0) atomicAdd(&counts[h], 1);
}

__global__ void k_scan(const int* __restrict__ counts, int* __restrict__ off) {
    if (threadIdx.x == 0 && blockIdx.x == 0) {
        int acc = 0;
        off[0] = 0;
        for (int i = 0; i < NHEAD; ++i) {
            acc += (counts[i] + 15) & ~15;   // 16-align each head's region
            off[i + 1] = acc;
        }
    }
}

__global__ void k_place(const int* __restrict__ head, const int* __restrict__ off,
                        int* __restrict__ cursors, int* __restrict__ list) {
    int t = blockIdx.x * blockDim.x + threadIdx.x;
    if (t >= NTOK) return;
    int h = head[t];
    if (h < 0) return;
    int p = atomicAdd(&cursors[h], 1);
    list[off[h] + p] = t;
}

// ---------------- weight convert + transpose (one-time per launch) ----------

// W1 [16][512][256] f32  ->  W1T [16][256][512] f16   (B^T: contiguous K)
__global__ void k_cvt_w1(const float* __restrict__ W1, _Float16* __restrict__ W1T) {
    int idx = blockIdx.x * blockDim.x + threadIdx.x;   // over 16*256*512 outputs
    if (idx >= NHEAD * HDIM * DDIM) return;
    int d = idx & (DDIM - 1);
    int j = (idx >> 9) & (HDIM - 1);
    int h = idx >> 17;
    W1T[idx] = (_Float16)W1[((size_t)h * DDIM + d) * HDIM + j];
}

// W2 [16][256][256] f32  ->  W2T [16][256][256] f16 transposed
__global__ void k_cvt_w2(const float* __restrict__ W2, _Float16* __restrict__ W2T) {
    int idx = blockIdx.x * blockDim.x + threadIdx.x;   // over 16*256*256 outputs
    if (idx >= NHEAD * HDIM * HDIM) return;
    int k = idx & (HDIM - 1);
    int j = (idx >> 8) & (HDIM - 1);
    int h = idx >> 16;
    W2T[idx] = (_Float16)W2[((size_t)h * HDIM + k) * HDIM + j];
}

// ---------------- GEMM1: relu(X @ W1 + b1), gathered rows, K=512 ------------

__global__ __launch_bounds__(256) void gemm1_kernel(
    const float* __restrict__ X,        // [8192, 512] f32
    const _Float16* __restrict__ W1T,   // [16][256][512] f16
    const float* __restrict__ b1,       // [16][256]
    const int* __restrict__ off,        // [17]
    const int* __restrict__ list,       // [LISTLEN]
    _Float16* __restrict__ H1)          // [LISTLEN][256] f16
{
    __shared__ _Float16 As[16 * DDIM];  // 16 KB: 16 gathered rows, f16
    int rowbase = blockIdx.x * 16;
    if (rowbase >= off[16]) return;     // uniform: whole block exits, EXEC stays full
    int h = 0;
    #pragma unroll
    for (int i = 1; i < NHEAD; ++i) if (off[i] <= rowbase) h = i;

    int tid = threadIdx.x;
    {   // cooperative gather + f32->f16 convert: 16 rows x (16 thr * 32 elems)
        int r = tid >> 4;
        int c = (tid & 15) * 32;
        int token = list[rowbase + r];
        _Float16* dst = &As[r * DDIM + c];
        if (token >= 0) {
            const float* src = X + (size_t)token * DDIM + c;
            #pragma unroll
            for (int i = 0; i < 32; i += 8) {
                float4 f0 = *(const float4*)(src + i);
                float4 f1 = *(const float4*)(src + i + 4);
                v8h hv;
                hv[0] = (_Float16)f0.x; hv[1] = (_Float16)f0.y;
                hv[2] = (_Float16)f0.z; hv[3] = (_Float16)f0.w;
                hv[4] = (_Float16)f1.x; hv[5] = (_Float16)f1.y;
                hv[6] = (_Float16)f1.z; hv[7] = (_Float16)f1.w;
                *(v8h*)(dst + i) = hv;              // ds_store_b128
            }
        } else {
            v8h z = {};
            #pragma unroll
            for (int i = 0; i < 32; i += 8) *(v8h*)(dst + i) = z;
        }
    }
    __syncthreads();

    int wave = tid >> 5, lane = tid & 31, lrow = lane & 15;
    bool hiHalf = lane >= 16;
    // two N-tiles per wave, A fragment shared per k-step
    int n0 = (wave * 2) * 16 + lrow;
    int n1 = n0 + 16;
    const _Float16* Bp0 = W1T + ((size_t)(h * HDIM + n0)) * DDIM;
    const _Float16* Bp1 = W1T + ((size_t)(h * HDIM + n1)) * DDIM;
    v8f acc0 = {}, acc1 = {};
    #pragma unroll
    for (int kb = 0; kb < DDIM / 32; ++kb) {
        v16h a, b0, b1v;
        // A layout: lane<16 -> K[0..15]; lane>=16 -> K[16..31]
        int ka = kb * 32 + (hiHalf ? 16 : 0);
        const _Float16* ap = &As[lrow * DDIM + ka];
        ((v8h*)&a)[0] = *(const v8h*)(ap);
        ((v8h*)&a)[1] = *(const v8h*)(ap + 8);
        // B layout: lane<16 -> K[0..15], lane>=16 -> K[16..31], col = lane%16
        int kbb = kb * 32 + (hiHalf ? 16 : 0);
        b0  = *(const v16h*)(Bp0 + kbb);
        b1v = *(const v16h*)(Bp1 + kbb);
        acc0 = __builtin_amdgcn_wmma_f32_16x16x32_f16(
            false, a, false, b0, (short)0, acc0, false, false);
        acc1 = __builtin_amdgcn_wmma_f32_16x16x32_f16(
            false, a, false, b1v, (short)0, acc1, false, false);
    }
    float bias0 = b1[h * HDIM + n0];
    float bias1 = b1[h * HDIM + n1];
    #pragma unroll
    for (int v = 0; v < 8; ++v) {             // C: row = v + (lane>=16 ? 8 : 0)
        int row = rowbase + v + (hiHalf ? 8 : 0);
        float v0 = acc0[v] + bias0; v0 = v0 > 0.0f ? v0 : 0.0f;
        float v1 = acc1[v] + bias1; v1 = v1 > 0.0f ? v1 : 0.0f;
        H1[(size_t)row * HDIM + n0] = (_Float16)v0;
        H1[(size_t)row * HDIM + n1] = (_Float16)v1;
    }
}

// ---------------- GEMM2: relu(H1 @ W2 + b2), contiguous rows, K=256 ---------

__global__ __launch_bounds__(256) void gemm2_kernel(
    const _Float16* __restrict__ H1,    // [LISTLEN][256] f16
    const _Float16* __restrict__ W2T,   // [16][256][256] f16
    const float* __restrict__ b2,       // [16][256]
    const int* __restrict__ off,
    _Float16* __restrict__ H2)          // [LISTLEN][256] f16
{
    int rowbase = blockIdx.x * 16;
    if (rowbase >= off[16]) return;
    int h = 0;
    #pragma unroll
    for (int i = 1; i < NHEAD; ++i) if (off[i] <= rowbase) h = i;

    int tid = threadIdx.x, wave = tid >> 5, lane = tid & 31, lrow = lane & 15;
    bool hiHalf = lane >= 16;
    const _Float16* Ap = H1 + (size_t)(rowbase + lrow) * HDIM;
    int n0 = (wave * 2) * 16 + lrow;
    int n1 = n0 + 16;
    const _Float16* Bp0 = W2T + ((size_t)(h * HDIM + n0)) * HDIM;
    const _Float16* Bp1 = W2T + ((size_t)(h * HDIM + n1)) * HDIM;
    v8f acc0 = {}, acc1 = {};
    #pragma unroll
    for (int kb = 0; kb < HDIM / 32; ++kb) {
        v16h a, b0, b1v;
        int ka = kb * 32 + (hiHalf ? 16 : 0);
        ((v8h*)&a)[0] = *(const v8h*)(Ap + ka);
        ((v8h*)&a)[1] = *(const v8h*)(Ap + ka + 8);
        int kbb = kb * 32 + (hiHalf ? 16 : 0);
        b0  = *(const v16h*)(Bp0 + kbb);
        b1v = *(const v16h*)(Bp1 + kbb);
        acc0 = __builtin_amdgcn_wmma_f32_16x16x32_f16(
            false, a, false, b0, (short)0, acc0, false, false);
        acc1 = __builtin_amdgcn_wmma_f32_16x16x32_f16(
            false, a, false, b1v, (short)0, acc1, false, false);
    }
    float bias0 = b2[h * HDIM + n0];
    float bias1 = b2[h * HDIM + n1];
    #pragma unroll
    for (int v = 0; v < 8; ++v) {
        int row = rowbase + v + (hiHalf ? 8 : 0);
        float v0 = acc0[v] + bias0; v0 = v0 > 0.0f ? v0 : 0.0f;
        float v1 = acc1[v] + bias1; v1 = v1 > 0.0f ? v1 : 0.0f;
        H2[(size_t)row * HDIM + n0] = (_Float16)v0;
        H2[(size_t)row * HDIM + n1] = (_Float16)v1;
    }
}

// ---------------- layer3 (H x 2) + scatter by token -------------------------

__global__ void k_out(const _Float16* __restrict__ H2,
                      const float* __restrict__ W3,   // [16][256][2]
                      const float* __restrict__ b3,   // [16][2]
                      const int* __restrict__ off, const int* __restrict__ list,
                      float* __restrict__ out) {
    int r = blockIdx.x * blockDim.x + threadIdx.x;
    if (r >= off[16]) return;
    int token = list[r];
    if (token < 0) return;                 // pad slot
    int h = 0;
    #pragma unroll
    for (int i = 1; i < NHEAD; ++i) if (off[i] <= r) h = i;
    const float* w = W3 + (size_t)h * HDIM * 2;
    float a0 = b3[h * 2 + 0], a1 = b3[h * 2 + 1];
    const _Float16* x = H2 + (size_t)r * HDIM;
    #pragma unroll 4
    for (int k = 0; k < HDIM; ++k) {
        float v = (float)x[k];
        a0 += v * w[k * 2 + 0];
        a1 += v * w[k * 2 + 1];
    }
    out[token * 2 + 0] = a0;
    out[token * 2 + 1] = a1;
}

// ---------------- host-side orchestration ------------------------------------

extern "C" void kernel_launch(void* const* d_in, const int* in_sizes, int n_in,
                              void* d_out, int out_size, void* d_ws, size_t ws_size,
                              hipStream_t stream) {
    const float* X    = (const float*)d_in[0];   // [32,256,512]
    const int*   tasks= (const int*)  d_in[1];   // [32,256]
    const int*   lut  = (const int*)  d_in[2];   // [1024]
    const float* W1   = (const float*)d_in[3];   // [16,512,256]
    const float* b1   = (const float*)d_in[4];   // [16,256]
    const float* W2   = (const float*)d_in[5];   // [16,256,256]
    const float* b2   = (const float*)d_in[6];   // [16,256]
    const float* W3   = (const float*)d_in[7];   // [16,256,2]
    const float* b3   = (const float*)d_in[8];   // [16,2]
    float* out = (float*)d_out;

    // workspace carve-up (~14.3 MB)
    char* ws = (char*)d_ws;
    size_t p = 0;
    int* head    = (int*)(ws + p); p += (size_t)NTOK * sizeof(int);          // 32768
    int* counts  = (int*)(ws + p); p += 64;
    int* cursors = (int*)(ws + p); p += 64;
    p = (p + 255) & ~(size_t)255;
    int* off     = (int*)(ws + p); p += 256;                                  // 17 ints
    int* list    = (int*)(ws + p); p += (size_t)LISTLEN * sizeof(int);
    p = (p + 255) & ~(size_t)255;
    _Float16* W1T = (_Float16*)(ws + p); p += (size_t)NHEAD * HDIM * DDIM * 2;
    _Float16* W2T = (_Float16*)(ws + p); p += (size_t)NHEAD * HDIM * HDIM * 2;
    _Float16* H1  = (_Float16*)(ws + p); p += (size_t)LISTLEN * HDIM * 2;
    _Float16* H2  = (_Float16*)(ws + p); p += (size_t)LISTLEN * HDIM * 2;
    (void)ws_size; (void)n_in; (void)in_sizes; (void)out_size;

    // weight convert/transpose to f16 (B^T layout for contiguous WMMA B-frags)
    k_cvt_w1<<<(NHEAD * HDIM * DDIM + 255) / 256, 256, 0, stream>>>(W1, W1T);
    k_cvt_w2<<<(NHEAD * HDIM * HDIM + 255) / 256, 256, 0, stream>>>(W2, W2T);

    // routing + bucketing
    k_init_counts<<<1, 32, 0, stream>>>(counts, cursors);
    k_route<<<(LISTLEN + 255) / 256, 256, 0, stream>>>(tasks, lut, head, counts, list, out);
    k_scan<<<1, 32, 0, stream>>>(counts, off);
    k_place<<<(NTOK + 255) / 256, 256, 0, stream>>>(head, off, cursors, list);

    // routed MLP: two WMMA GEMMs + skinny output layer with scatter
    gemm1_kernel<<<MTILES, 256, 0, stream>>>(X, W1T, b1, off, list, H1);
    gemm2_kernel<<<MTILES, 256, 0, stream>>>(H1, W2T, b2, off, H2);
    k_out<<<(LISTLEN + 255) / 256, 256, 0, stream>>>(H2, W3, b3, off, list, out);
}
